// Model_489626272354
// MI455X (gfx1250) — matrix operand order repaired and checked by reading, not compiled, not run on hardware
//
#include <hip/hip_runtime.h>
#include <math.h>

typedef __attribute__((ext_vector_type(16))) _Float16 v16h;
typedef __attribute__((ext_vector_type(8)))  float    v8f;
typedef __attribute__((ext_vector_type(4)))  unsigned int v4u;
typedef __attribute__((ext_vector_type(8)))  int      v8i;
typedef __attribute__((ext_vector_type(4)))  int      v4i;

#define HID    40
#define NG     160          // 4*HID gates
#define NOUT   21           // gamma + 20 eta
#define TSTEPS 192
#define BATCH  4096
#define SEQ    96
#define FRAW   20
#define FCONV  22
#define FCNN   11
#define K0P    64           // cell0 K = 11+40=51 -> pad 64
#define K1P    96           // cell1/head K = 80 -> pad 96

#if defined(__has_builtin)
#if __has_builtin(__builtin_amdgcn_tensor_load_to_lds) && \
    __has_builtin(__builtin_amdgcn_s_wait_tensorcnt)
#define USE_TDM 1
#endif
#endif
#ifndef USE_TDM
#define USE_TDM 0
#endif

__device__ __forceinline__ float sigm(float x) { return 1.0f / (1.0f + __expf(-x)); }
__device__ __forceinline__ float softplus(float x) {
    return fmaxf(x, 0.0f) + log1pf(__expf(-fabsf(x)));
}

#if USE_TDM
// TDM descriptor (ISA 8.3/8.4): 2D tile 16 x 11 f32, row stride 11 f32,
// contiguous into LDS at lds_off.  All fields wave-uniform.
// 6-arg builtin flavor (clang-23 / therock-10.0): extra zero-filled v8i group.
__device__ __forceinline__ void tdm_load_feat(const float* gsrc, unsigned lds_off) {
    unsigned long long ga = (unsigned long long)(size_t)gsrc;
    v4u g0;
    g0[0] = 1u;                                   // count=1, load, user mode
    g0[1] = lds_off;                              // lds_addr (bytes)
    g0[2] = (unsigned)ga;                         // global_addr[31:0]
    g0[3] = (unsigned)((ga >> 32) & 0x01FFFFFFull) | (2u << 30);  // addr[56:32] | type=2
    v8i g1;
    g1[0] = (2 << 16);                            // data_size = 4 bytes
    g1[1] = (FCNN << 16);                         // tensor_dim0 = 11
    g1[2] = (16 << 16);                           // tensor_dim1 = 16
    g1[3] = (FCNN << 16);                         // tile_dim0 = 11
    g1[4] = 16;                                   // tile_dim1 = 16 (tile_dim2 = 0)
    g1[5] = FCNN;                                 // tensor_dim0_stride = 11
    g1[6] = 0;
    g1[7] = 0;
    v4i gz4 = {0, 0, 0, 0};                       // 2D tensor: groups 2/3 unused
    v8i gz8 = {0, 0, 0, 0, 0, 0, 0, 0};           // extra group (6-arg flavor), zero
    __builtin_amdgcn_tensor_load_to_lds(g0, g1, gz4, gz4, gz8, 0);
}
#endif

// ---------------------------------------------------------------------------
// Kernel 1: conv(k=3, circular pad 2) + training-mode BN over (B, L) + ELU +
// maxpool(3,2,1).  One block per timestep t; block-wide reduction for BN.
// feat layout: [T][B][FCNN] f32 in workspace.
// ---------------------------------------------------------------------------
__global__ __launch_bounds__(256) void conv_bn_pool_kernel(
    const float* __restrict__ x_enc, const float* __restrict__ y_enc,
    const float* __restrict__ conv_w, const float* __restrict__ conv_b,
    const float* __restrict__ bn_w, const float* __restrict__ bn_b,
    float* __restrict__ feat)
{
    const int t   = blockIdx.x;
    const int tid = threadIdx.x;
    const float w0 = conv_w[0], w1 = conv_w[1], w2 = conv_w[2], cb = conv_b[0];
    const float* src = (t < SEQ) ? x_enc : y_enc;
    const int    tt  = (t < SEQ) ? t : (t - SEQ);

    float s = 0.0f, ss = 0.0f;
    for (int b = tid; b < BATCH; b += 256) {
        const float* xr = src + ((size_t)b * SEQ + tt) * (FRAW + 1);
        float x[FRAW];
        #pragma unroll
        for (int f = 0; f < FRAW; ++f) x[f] = xr[f];
        #pragma unroll
        for (int l = 0; l < FCONV; ++l) {
            float y = w0 * x[(l + 18) % FRAW] + w1 * x[(l + 19) % FRAW]
                    + w2 * x[l % FRAW] + cb;
            s += y; ss += y * y;
        }
    }

    __shared__ float red[256], red2[256];
    red[tid] = s; red2[tid] = ss;
    __syncthreads();
    for (int off = 128; off > 0; off >>= 1) {
        if (tid < off) { red[tid] += red[tid + off]; red2[tid] += red2[tid + off]; }
        __syncthreads();
    }
    __shared__ float s_mu, s_inv;
    if (tid == 0) {
        const float cnt = (float)(BATCH * FCONV);
        float mu  = red[0] / cnt;
        float var = red2[0] / cnt - mu * mu;
        s_mu = mu; s_inv = rsqrtf(var + 1e-5f);
    }
    __syncthreads();
    const float mu = s_mu, inv = s_inv, g = bn_w[0], bb = bn_b[0];

    for (int b = tid; b < BATCH; b += 256) {
        const float* xr = src + ((size_t)b * SEQ + tt) * (FRAW + 1);
        float x[FRAW];
        #pragma unroll
        for (int f = 0; f < FRAW; ++f) x[f] = xr[f];
        float y[FCONV];
        #pragma unroll
        for (int l = 0; l < FCONV; ++l) {
            float v = w0 * x[(l + 18) % FRAW] + w1 * x[(l + 19) % FRAW]
                    + w2 * x[l % FRAW] + cb;
            v = (v - mu) * inv * g + bb;
            y[l] = (v > 0.0f) ? v : (__expf(v) - 1.0f);   // ELU
        }
        float* fr = feat + ((size_t)t * BATCH + b) * FCNN;
        #pragma unroll
        for (int j = 0; j < FCNN; ++j) {                   // maxpool k=3,s=2,p=1
            float m = y[2 * j];
            if (j > 0) m = fmaxf(m, y[2 * j - 1]);
            m = fmaxf(m, y[2 * j + 1]);                    // 2j+1 <= 21 < 22
            fr[j] = m;
        }
    }
}

// ---------------------------------------------------------------------------
// Kernel 2: recurrent 2-layer LSTM + heads, WMMA f16->f32.
// 2 waves per WG, each wave owns 16 batch rows for all 192 steps.
// Feature tiles stream in via the Tensor Data Mover (double-buffered,
// TENSORcnt-pipelined); weights live in LDS as f16, K-concatenated:
//   Bc0[n][k] : k<11 w_ih0, k<51 w_hh0           (K0P=64)
//   Bc1[n][k] : k<40 w_ih1, k<80 w_hh1           (K1P=96)
//   Bhd[n][k] : head weights, columns permuted to de-interleave hp
// ---------------------------------------------------------------------------
__global__ __launch_bounds__(64) void lstm_wmma_kernel(
    const float* __restrict__ feat,
    const float* __restrict__ w_ih0, const float* __restrict__ w_hh0,
    const float* __restrict__ b0,
    const float* __restrict__ w_ih1, const float* __restrict__ w_hh1,
    const float* __restrict__ b1,
    const float* __restrict__ w_gamma, const float* __restrict__ b_gamma,
    const float* __restrict__ w_eta,   const float* __restrict__ b_eta,
    float* __restrict__ out)
{
    __shared__ _Float16 Bc0[NG][K0P];
    __shared__ _Float16 Bc1[NG][K1P];
    __shared__ _Float16 Bhd[32][K1P];
    __shared__ float    bs0[NG], bs1[NG], bsh[32];
    __shared__ _Float16 xb0[2][16][K0P];   // [feat | h0 | 0]
    __shared__ _Float16 xb1[2][16][K1P];   // [h0 | h1 | 0]
    __shared__ float    gb[2][16][NG];     // gate accumulators
#if USE_TDM
    __shared__ float    fbuf[2][2][16][FCNN];   // [wave][slot][m][j]
#endif

    const int tid = threadIdx.x;

    for (int idx = tid; idx < NG * K0P; idx += 64) {
        int n = idx / K0P, k = idx % K0P;
        float v = (k < FCNN) ? w_ih0[n * FCNN + k]
                : (k < FCNN + HID) ? w_hh0[n * HID + (k - FCNN)] : 0.0f;
        Bc0[n][k] = (_Float16)v;
    }
    for (int idx = tid; idx < NG * K1P; idx += 64) {
        int n = idx / K1P, k = idx % K1P;
        float v = (k < HID) ? w_ih1[n * HID + k]
                : (k < 2 * HID) ? w_hh1[n * HID + (k - HID)] : 0.0f;
        Bc1[n][k] = (_Float16)v;
    }
    for (int idx = tid; idx < 32 * K1P; idx += 64) {
        int n = idx / K1P, k = idx % K1P;
        float v = 0.0f;
        if (k < 2 * HID && n < NOUT) {
            int j  = (k < HID) ? k : (k - HID);
            int kk = (k < HID) ? (2 * j) : (2 * j + 1);     // de-interleave hp
            v = (n == 0) ? w_gamma[kk] : w_eta[(n - 1) * 2 * HID + kk];
        }
        Bhd[n][k] = (_Float16)v;
    }
    for (int idx = tid; idx < NG; idx += 64) { bs0[idx] = b0[idx]; bs1[idx] = b1[idx]; }
    if (tid < 32) bsh[tid] = (tid == 0) ? b_gamma[0] : (tid < NOUT ? b_eta[tid - 1] : 0.0f);

    const int wv   = tid >> 5;
    const int lane = tid & 31;
    for (int idx = lane; idx < 16 * K0P; idx += 32) (&xb0[wv][0][0])[idx] = (_Float16)0.0f;
    for (int idx = lane; idx < 16 * K1P; idx += 32) (&xb1[wv][0][0])[idx] = (_Float16)0.0f;
    __syncthreads();

    const int m0 = blockIdx.x * 32 + wv * 16;   // batch row base of this wave
    const int hi = lane >> 4;
    const int nl = lane & 15;

    float c0reg[20], c1reg[20];
    #pragma unroll
    for (int i = 0; i < 20; ++i) { c0reg[i] = 0.0f; c1reg[i] = 0.0f; }

#if USE_TDM
    // prime the pipeline: DMA feat tile for t=0 into slot 0
    tdm_load_feat(feat + (size_t)m0 * FCNN,
                  (unsigned)(size_t)&fbuf[wv][0][0][0]);
#endif

    for (int t = 0; t < TSTEPS; ++t) {
        // ---- 1) stage feat tile into xb0[:,0..10] -------------------------
#if USE_TDM
        __builtin_amdgcn_s_wait_tensorcnt(0);          // slot t&1 ready
        if (t + 1 < TSTEPS)                            // overlap next DMA with compute
            tdm_load_feat(feat + ((size_t)(t + 1) * BATCH + m0) * FCNN,
                          (unsigned)(size_t)&fbuf[wv][(t + 1) & 1][0][0]);
        {
            const float* fb = &fbuf[wv][t & 1][0][0];
            for (int idx = lane; idx < 16 * FCNN; idx += 32)
                (&xb0[wv][0][0])[(idx / FCNN) * K0P + (idx % FCNN)] = (_Float16)fb[idx];
        }
#else
        for (int idx = lane; idx < 16 * FCNN; idx += 32) {
            int m = idx / FCNN, j = idx % FCNN;
            xb0[wv][m][j] = (_Float16)feat[((size_t)t * BATCH + (m0 + m)) * FCNN + j];
        }
#endif
        if (t + 2 < TSTEPS && lane < 16)
            __builtin_prefetch(&feat[((size_t)(t + 2) * BATCH + (m0 + lane)) * FCNN], 0, 1);
        asm volatile("s_wait_dscnt 0" ::: "memory");
        __builtin_amdgcn_wave_barrier();

        // ---- 2) gates0 = [feat|h0] x Bc0^T  (K=64, 2 frags, 10 N-tiles) ---
        #pragma unroll
        for (int nt = 0; nt < 10; ++nt) {
            v8f acc = {};
            #pragma unroll
            for (int kf = 0; kf < 2; ++kf) {
                const int kb = kf * 32;
                v16h a, b;
                const _Float16* pa0 = &xb0[wv][nl][kb + hi * 8];
                const _Float16* pa1 = &xb0[wv][nl][kb + 16 + hi * 8];
                #pragma unroll
                for (int e = 0; e < 8; ++e) { a[e] = pa0[e]; a[8 + e] = pa1[e]; }
                const _Float16* pb0 = &Bc0[nt * 16 + nl][kb + hi * 8];
                const _Float16* pb1 = &Bc0[nt * 16 + nl][kb + 16 + hi * 8];
                #pragma unroll
                for (int e = 0; e < 8; ++e) { b[e] = pb0[e]; b[8 + e] = pb1[e]; }
                acc = __builtin_amdgcn_wmma_f32_16x16x32_f16(
                        false, a, false, b, (short)0, acc, false, false);
            }
            #pragma unroll
            for (int r = 0; r < 8; ++r) gb[wv][r + hi * 8][nt * 16 + nl] = acc[r];
        }
        asm volatile("s_wait_dscnt 0" ::: "memory");
        __builtin_amdgcn_wave_barrier();

        // ---- 3) cell0 elementwise (i,f,g,o), state in registers ----------
        #pragma unroll
        for (int ii = 0; ii < 20; ++ii) {
            int cell = ii * 32 + lane;
            int m = cell / HID, j = cell % HID;
            float gi = gb[wv][m][j]           + bs0[j];
            float gf = gb[wv][m][HID + j]     + bs0[HID + j];
            float gg = gb[wv][m][2 * HID + j] + bs0[2 * HID + j];
            float go = gb[wv][m][3 * HID + j] + bs0[3 * HID + j];
            float c  = sigm(gf) * c0reg[ii] + sigm(gi) * tanhf(gg);
            c0reg[ii] = c;
            float h  = sigm(go) * tanhf(c);
            _Float16 hh = (_Float16)h;
            xb0[wv][m][FCNN + j] = hh;     // recurrent input for next step
            xb1[wv][m][j]        = hh;     // input to cell1 / head
        }
        asm volatile("s_wait_dscnt 0" ::: "memory");
        __builtin_amdgcn_wave_barrier();

        // ---- 4) gates1 = [h0|h1] x Bc1^T  (K=96, 3 frags, 10 N-tiles) ----
        #pragma unroll
        for (int nt = 0; nt < 10; ++nt) {
            v8f acc = {};
            #pragma unroll
            for (int kf = 0; kf < 3; ++kf) {
                const int kb = kf * 32;
                v16h a, b;
                const _Float16* pa0 = &xb1[wv][nl][kb + hi * 8];
                const _Float16* pa1 = &xb1[wv][nl][kb + 16 + hi * 8];
                #pragma unroll
                for (int e = 0; e < 8; ++e) { a[e] = pa0[e]; a[8 + e] = pa1[e]; }
                const _Float16* pb0 = &Bc1[nt * 16 + nl][kb + hi * 8];
                const _Float16* pb1 = &Bc1[nt * 16 + nl][kb + 16 + hi * 8];
                #pragma unroll
                for (int e = 0; e < 8; ++e) { b[e] = pb0[e]; b[8 + e] = pb1[e]; }
                acc = __builtin_amdgcn_wmma_f32_16x16x32_f16(
                        false, a, false, b, (short)0, acc, false, false);
            }
            #pragma unroll
            for (int r = 0; r < 8; ++r) gb[wv][r + hi * 8][nt * 16 + nl] = acc[r];
        }
        asm volatile("s_wait_dscnt 0" ::: "memory");
        __builtin_amdgcn_wave_barrier();

        // ---- 5) cell1 elementwise ----------------------------------------
        #pragma unroll
        for (int ii = 0; ii < 20; ++ii) {
            int cell = ii * 32 + lane;
            int m = cell / HID, j = cell % HID;
            float gi = gb[wv][m][j]           + bs1[j];
            float gf = gb[wv][m][HID + j]     + bs1[HID + j];
            float gg = gb[wv][m][2 * HID + j] + bs1[2 * HID + j];
            float go = gb[wv][m][3 * HID + j] + bs1[3 * HID + j];
            float c  = sigm(gf) * c1reg[ii] + sigm(gi) * tanhf(gg);
            c1reg[ii] = c;
            float h  = sigm(go) * tanhf(c);
            xb1[wv][m][HID + j] = (_Float16)h;
        }
        asm volatile("s_wait_dscnt 0" ::: "memory");
        __builtin_amdgcn_wave_barrier();

        // ---- 6) heads: softplus([h0|h1] x Bhd^T + b), direct global store
        #pragma unroll
        for (int nt = 0; nt < 2; ++nt) {
            v8f acc = {};
            #pragma unroll
            for (int kf = 0; kf < 3; ++kf) {
                const int kb = kf * 32;
                v16h a, b;
                const _Float16* pa0 = &xb1[wv][nl][kb + hi * 8];
                const _Float16* pa1 = &xb1[wv][nl][kb + 16 + hi * 8];
                #pragma unroll
                for (int e = 0; e < 8; ++e) { a[e] = pa0[e]; a[8 + e] = pa1[e]; }
                const _Float16* pb0 = &Bhd[nt * 16 + nl][kb + hi * 8];
                const _Float16* pb1 = &Bhd[nt * 16 + nl][kb + 16 + hi * 8];
                #pragma unroll
                for (int e = 0; e < 8; ++e) { b[e] = pb0[e]; b[8 + e] = pb1[e]; }
                acc = __builtin_amdgcn_wmma_f32_16x16x32_f16(
                        false, a, false, b, (short)0, acc, false, false);
            }
            const int n = nt * 16 + nl;
            if (n < NOUT) {
                #pragma unroll
                for (int r = 0; r < 8; ++r) {
                    int m = r + hi * 8;
                    float v = softplus(acc[r] + bsh[n]);
                    out[((size_t)t * BATCH + (m0 + m)) * NOUT + n] = v;
                }
            }
        }
    }
}

extern "C" void kernel_launch(void* const* d_in, const int* in_sizes, int n_in,
                              void* d_out, int out_size, void* d_ws, size_t ws_size,
                              hipStream_t stream) {
    const float* x_enc   = (const float*)d_in[0];
    const float* y_enc   = (const float*)d_in[3];
    const float* conv_w  = (const float*)d_in[5];
    const float* conv_b  = (const float*)d_in[6];
    const float* bn_w    = (const float*)d_in[7];
    const float* bn_b    = (const float*)d_in[8];
    const float* w_ih0   = (const float*)d_in[9];
    const float* w_hh0   = (const float*)d_in[10];
    const float* b0      = (const float*)d_in[11];
    const float* w_ih1   = (const float*)d_in[12];
    const float* w_hh1   = (const float*)d_in[13];
    const float* b1      = (const float*)d_in[14];
    const float* w_gamma = (const float*)d_in[15];
    const float* b_gamma = (const float*)d_in[16];
    const float* w_eta   = (const float*)d_in[17];
    const float* b_eta   = (const float*)d_in[18];

    float* feat = (float*)d_ws;   // [T][B][FCNN] f32, ~33 MB

    conv_bn_pool_kernel<<<TSTEPS, 256, 0, stream>>>(
        x_enc, y_enc, conv_w, conv_b, bn_w, bn_b, feat);

    lstm_wmma_kernel<<<BATCH / 32, 64, 0, stream>>>(
        feat, w_ih0, w_hh0, b0, w_ih1, w_hh1, b1,
        w_gamma, b_gamma, w_eta, b_eta, (float*)d_out);
}
